// CrossAttention_24816321036603
// MI455X (gfx1250) — matrix operand order repaired and checked by reading, not compiled, not run on hardware
//
#include <hip/hip_runtime.h>
#include <stdint.h>

typedef __attribute__((ext_vector_type(16))) _Float16 v16h;
typedef __attribute__((ext_vector_type(8)))  float    v8f;

#define B_      4
#define C_      256
#define NPIX    4096
#define NH_     4
#define HD_     64
#define GROUPS_ 32
#define EPS_    1e-5f

static __device__ __forceinline__ v8f wmma_f16(v16h a, v16h b, v8f c) {
  // D = A(16x32 f16) * B(32x16 f16) + C(16x16 f32)
  return __builtin_amdgcn_wmma_f32_16x16x32_f16(false, a, false, b, (short)0, c,
                                                false, false);
}

// A-fragment (16x32 f16): per-lane element i (0..15) -> K index; one
// contiguous 16-element run at 16*lh  => 2x b128 loads.
static __device__ __forceinline__ int a_kmap(int i, int lh) {
  return 16 * lh + i;
}

// CDNA5 async DMA: memory -> LDS, 16 bytes per lane, tracked by ASYNCcnt.
// LDS byte address = low 32 bits of the flat shared pointer (ISA 10.2:
// LDS aperture uses addr[31:0]).
static __device__ __forceinline__ void async_b128(const void* g, void* lds) {
  unsigned l = (unsigned)(uintptr_t)lds;
  unsigned long long a = (unsigned long long)(uintptr_t)g;
  asm volatile("global_load_async_to_lds_b128 %0, %1, off"
               :: "v"(l), "v"(a) : "memory");
}
static __device__ __forceinline__ void wait_async0() {
  asm volatile("s_wait_asynccnt 0x0" ::: "memory");
}

// ---------------------------------------------------------------------------
// Weights f32 -> f16
// ---------------------------------------------------------------------------
__global__ void f32_to_f16_kernel(const float* __restrict__ src,
                                  _Float16* __restrict__ dst, int n) {
  int i = blockIdx.x * blockDim.x + threadIdx.x;
  if (i < n) dst[i] = (_Float16)src[i];
}

// ---------------------------------------------------------------------------
// GroupNorm: one block per (batch, group); 8 channels x 4096 pix.
// Reads channel-major input, writes PIXEL-MAJOR f16 activations [b][n][c].
// ---------------------------------------------------------------------------
__global__ void __launch_bounds__(256)
groupnorm_f16_kernel(const float* __restrict__ x,
                     const float* __restrict__ gamma,
                     const float* __restrict__ beta,
                     _Float16* __restrict__ out) {
  const int b = blockIdx.x / GROUPS_;
  const int g = blockIdx.x % GROUPS_;
  const int CPG  = C_ / GROUPS_;    // 8
  const int SPAN = CPG * NPIX;      // 32768
  const size_t base = ((size_t)b * C_ + (size_t)g * CPG) * NPIX;
  const int t = threadIdx.x;

  __shared__ float s_sum[256];
  __shared__ float s_sq[256];

  float sum = 0.f, sq = 0.f;
  const float4* xv = (const float4*)(x + base);
  for (int i = t; i < SPAN / 4; i += 256) {
    float4 v = xv[i];
    sum += v.x + v.y + v.z + v.w;
    sq  += v.x * v.x + v.y * v.y + v.z * v.z + v.w * v.w;
  }
  s_sum[t] = sum;
  s_sq[t]  = sq;
  __syncthreads();
  for (int off = 128; off > 0; off >>= 1) {
    if (t < off) { s_sum[t] += s_sum[t + off]; s_sq[t] += s_sq[t + off]; }
    __syncthreads();
  }
  const float mean = s_sum[0] / (float)SPAN;
  const float var  = s_sq[0] / (float)SPAN - mean * mean;
  const float rstd = rsqrtf(var + EPS_);

  for (int i = t; i < SPAN / 4; i += 256) {
    float4 v = xv[i];
    int cl = (i * 4) >> 12;          // channel within group
    int n  = (i * 4) & (NPIX - 1);   // pixel
    int c  = g * CPG + cl;
    float ga = gamma[c] * rstd;
    float bb = beta[c] - mean * ga;
    _Float16* o = out + ((size_t)b * NPIX + n) * C_ + c;  // pixel-major
    o[0 * C_] = (_Float16)(v.x * ga + bb);
    o[1 * C_] = (_Float16)(v.y * ga + bb);
    o[2 * C_] = (_Float16)(v.z * ga + bb);
    o[3 * C_] = (_Float16)(v.w * ga + bb);
  }
}

// ---------------------------------------------------------------------------
// WMMA GEMM: Out[b][n][m] = sum_k Act[b][n][k] * W[m][k] + bias[m]
// A = activation tile (pixels), B = W^T (out-channels); both tiles are
// K-contiguous in LDS => all fragment reads are b128.  Tiles are filled with
// GLOBAL_LOAD_ASYNC_TO_LDS_B128 and double-buffered: next K-step's DMA
// overlaps this K-step's WMMAs.
// mode 0: write f16 pixel-major.  mode 1: write f32 channel-major + residual.
// ---------------------------------------------------------------------------
__global__ void __launch_bounds__(256)
gemm_wmma_kernel(const _Float16* __restrict__ W,
                 const _Float16* __restrict__ X,   // pixel-major [b][n][k]
                 const float* __restrict__ bias,
                 _Float16* __restrict__ outh,
                 const float* __restrict__ resid,
                 float* __restrict__ outf,
                 int mode) {
  const int b  = blockIdx.z;
  const int n0 = blockIdx.x * 128;   // pixel tile
  const int m0 = blockIdx.y * 64;    // out-channel tile
  const int t  = threadIdx.x;
  const int wave = t >> 5, lane = t & 31;
  const int lh = lane >> 4, ln = lane & 15;

  __shared__ _Float16 As[2][128 * 32];  // [pixel][k]
  __shared__ _Float16 Ws[2][64 * 32];   // [m][k]

  // per-thread DMA source/dest assignments
  const int arow = t >> 1, ahf = t & 1;              // Act: 2 x b128 / thread
  const int wrow = t >> 2, wcg = t & 3;              // W:   1 x b128 / thread
  const _Float16* aSrc = &X[((size_t)b * NPIX + n0 + arow) * C_ + ahf * 16];
  const _Float16* wSrc = &W[(size_t)(m0 + wrow) * C_ + wcg * 8];
  const int aOff = arow * 32 + ahf * 16;
  const int wOff = wrow * 32 + wcg * 8;

  v8f acc[4];
#pragma unroll
  for (int j = 0; j < 4; ++j)
#pragma unroll
    for (int r = 0; r < 8; ++r) acc[j][r] = 0.f;

  // prologue: DMA K-step 0 into buffer 0
  async_b128(aSrc,     &As[0][aOff]);
  async_b128(aSrc + 8, &As[0][aOff + 8]);
  async_b128(wSrc,     &Ws[0][wOff]);

  const int KSTEPS = C_ / 32;  // 8
  for (int it = 0; it < KSTEPS; ++it) {
    const int cur = it & 1;
    wait_async0();      // my DMAs into buf[cur] are complete
    __syncthreads();    // everyone's are; buf[cur^1] no longer being read
    if (it + 1 < KSTEPS) {
      const int kk = (it + 1) * 32;
      async_b128(aSrc + kk,     &As[cur ^ 1][aOff]);
      async_b128(aSrc + kk + 8, &As[cur ^ 1][aOff + 8]);
      async_b128(wSrc + kk,     &Ws[cur ^ 1][wOff]);
    }

    v16h a;  // pixels 16*wave .. +16
#pragma unroll
    for (int i = 0; i < 16; ++i)
      a[i] = As[cur][(16 * wave + ln) * 32 + a_kmap(i, lh)];

#pragma unroll
    for (int j = 0; j < 4; ++j) {
      v16h bb;  // B-frag: column = out-channel, K contiguous
#pragma unroll
      for (int i = 0; i < 16; ++i)
        bb[i] = Ws[cur][(16 * j + ln) * 32 + 16 * lh + i];
      acc[j] = wmma_f16(a, bb, acc[j]);
    }
  }

  if (mode == 0) {
#pragma unroll
    for (int j = 0; j < 4; ++j)
#pragma unroll
      for (int r = 0; r < 8; ++r) {
        int pixel = n0 + 16 * wave + r + 8 * lh;
        int chan  = m0 + 16 * j + ln;
        outh[((size_t)b * NPIX + pixel) * C_ + chan] =
            (_Float16)(acc[j][r] + bias[chan]);
      }
  } else {
#pragma unroll
    for (int j = 0; j < 4; ++j) {
      int chan = m0 + 16 * j + ln;
      float bs = bias[chan];
      size_t idx = ((size_t)b * C_ + chan) * NPIX + n0 + 16 * wave + 8 * lh;
      float4 r0 = *(const float4*)(&resid[idx]);
      float4 r1 = *(const float4*)(&resid[idx + 4]);
      float4 o0, o1;
      o0.x = r0.x + acc[j][0] + bs; o0.y = r0.y + acc[j][1] + bs;
      o0.z = r0.z + acc[j][2] + bs; o0.w = r0.w + acc[j][3] + bs;
      o1.x = r1.x + acc[j][4] + bs; o1.y = r1.y + acc[j][5] + bs;
      o1.z = r1.z + acc[j][6] + bs; o1.w = r1.w + acc[j][7] + bs;
      *(float4*)(&outf[idx])     = o0;
      *(float4*)(&outf[idx + 4]) = o1;
    }
  }
}

// ---------------------------------------------------------------------------
// Flash attention per (b, h). Q,K,V pixel-major [b][n][256] f16.
// Block: 128 queries (8 waves x 16); keys streamed in chunks of 32.
// K tile: async DMA, double-buffered.  V tile: register-prefetched one chunk
// ahead, scattered (transposed) into LDS before the barrier.
// ---------------------------------------------------------------------------
__global__ void __launch_bounds__(256)
attn_kernel(const _Float16* __restrict__ qh,
            const _Float16* __restrict__ kh,
            const _Float16* __restrict__ vh,
            _Float16* __restrict__ oh) {
  const int b  = blockIdx.z;
  const int h  = blockIdx.y;
  const int n0 = blockIdx.x * 128;
  const int t  = threadIdx.x;
  const int wave = t >> 5, lane = t & 31;
  const int lh = lane >> 4, ln = lane & 15;
  const int c0 = h * HD_;

  __shared__ _Float16 Ks[2][32 * 64];      // [key][d]   (native, via DMA)
  __shared__ _Float16 Vs[2][64 * 32];      // [d][key]   (transposed scatter)
  __shared__ _Float16 p_lds[8 * 16 * 32];  // per-wave P [q][key]

  const size_t pbase = (size_t)b * NPIX;

  union U8 { uint4 u; _Float16 f[8]; };

  // Q A-fragments straight from global: row = query, K = d (contiguous)
  v16h aQ0, aQ1;
  {
    const _Float16* qrow = &qh[(pbase + n0 + 16 * wave + ln) * C_ + c0];
#pragma unroll
    for (int i = 0; i < 16; ++i) {
      int k = a_kmap(i, lh);
      aQ0[i] = qrow[k];
      aQ1[i] = qrow[32 + k];
    }
  }

  float mrow[8], lrow[8];
  v8f oacc[4];
#pragma unroll
  for (int r = 0; r < 8; ++r) { mrow[r] = -3.0e38f; lrow[r] = 0.f; }
#pragma unroll
  for (int j = 0; j < 4; ++j)
#pragma unroll
    for (int r = 0; r < 8; ++r) oacc[j][r] = 0.f;

  const float scale = 0.125f;  // hd^-0.5

  // per-thread K/V staging assignment: 8 f16 of one key row
  const int key = t >> 3, seg = t & 7;
  const _Float16* kSrc = &kh[(pbase + key) * C_ + c0 + seg * 8];
  const _Float16* vSrc = &vh[(pbase + key) * C_ + c0 + seg * 8];
  const int kOff = key * 64 + seg * 8;

  // prologue: chunk 0
  async_b128(kSrc, &Ks[0][kOff]);
  U8 vreg, vnext;
  vreg.u = *(const uint4*)(vSrc);
  vnext  = vreg;

  const int CHUNKS = NPIX / 32;  // 128
  for (int it = 0; it < CHUNKS; ++it) {
    const int cur = it & 1;
    wait_async0();  // K DMA into Ks[cur] complete (this wave)
    // scatter prefetched V chunk into Vs[cur] (transpose)
#pragma unroll
    for (int e = 0; e < 8; ++e) Vs[cur][(seg * 8 + e) * 32 + key] = vreg.f[e];
    __syncthreads();  // tiles[cur] ready everywhere; tiles[cur^1] free
    if (it + 1 < CHUNKS) {
      const size_t goff = (size_t)(it + 1) * 32 * C_;
      async_b128(kSrc + goff, &Ks[cur ^ 1][kOff]);
      vnext.u = *(const uint4*)(vSrc + goff);
    }

    // S tiles: s0 = keys [kt, kt+16), s1 = keys [kt+16, kt+32)
    v8f s0, s1;
#pragma unroll
    for (int r = 0; r < 8; ++r) { s0[r] = 0.f; s1[r] = 0.f; }
    {
      v16h bk;  // column = key, elements d contiguous
#pragma unroll
      for (int i = 0; i < 16; ++i) bk[i] = Ks[cur][ln * 64 + 16 * lh + i];
      s0 = wmma_f16(aQ0, bk, s0);
#pragma unroll
      for (int i = 0; i < 16; ++i) bk[i] = Ks[cur][ln * 64 + 32 + 16 * lh + i];
      s0 = wmma_f16(aQ1, bk, s0);
#pragma unroll
      for (int i = 0; i < 16; ++i) bk[i] = Ks[cur][(16 + ln) * 64 + 16 * lh + i];
      s1 = wmma_f16(aQ0, bk, s1);
#pragma unroll
      for (int i = 0; i < 16; ++i) bk[i] = Ks[cur][(16 + ln) * 64 + 32 + 16 * lh + i];
      s1 = wmma_f16(aQ1, bk, s1);
    }

    // Online softmax; row (r, lh) spans 16 lanes of one half.
    float p0[8], p1[8], alpha[8];
#pragma unroll
    for (int r = 0; r < 8; ++r) {
      float a_ = s0[r] * scale;
      float c_ = s1[r] * scale;
      float mx = fmaxf(a_, c_);
#pragma unroll
      for (int msk = 1; msk < 16; msk <<= 1)
        mx = fmaxf(mx, __shfl_xor(mx, msk, 32));
      float mn = fmaxf(mrow[r], mx);
      alpha[r] = __expf(mrow[r] - mn);
      mrow[r]  = mn;
      p0[r] = __expf(a_ - mn);
      p1[r] = __expf(c_ - mn);
      float rs = p0[r] + p1[r];
#pragma unroll
      for (int msk = 1; msk < 16; msk <<= 1)
        rs += __shfl_xor(rs, msk, 32);
      lrow[r] = lrow[r] * alpha[r] + rs;
    }
#pragma unroll
    for (int j = 0; j < 4; ++j)
#pragma unroll
      for (int r = 0; r < 8; ++r) oacc[j][r] *= alpha[r];

    // Restage P (C-layout) -> per-wave LDS [16 q][32 keys] f16
    _Float16* pw = &p_lds[wave * 512];
#pragma unroll
    for (int r = 0; r < 8; ++r) {
      pw[(r + 8 * lh) * 32 + ln]      = (_Float16)p0[r];
      pw[(r + 8 * lh) * 32 + 16 + ln] = (_Float16)p1[r];
    }
    asm volatile("s_wait_dscnt 0x0" ::: "memory");  // per-wave LDS RAW fence

    v16h aP;
#pragma unroll
    for (int i = 0; i < 16; ++i) aP[i] = pw[ln * 32 + a_kmap(i, lh)];

#pragma unroll
    for (int j = 0; j < 4; ++j) {
      v16h bv;  // column = d, elements key contiguous
#pragma unroll
      for (int i = 0; i < 16; ++i)
        bv[i] = Vs[cur][(16 * j + ln) * 32 + 16 * lh + i];
      oacc[j] = wmma_f16(aP, bv, oacc[j]);
    }
    vreg = vnext;
  }

#pragma unroll
  for (int r = 0; r < 8; ++r) lrow[r] = 1.f / lrow[r];
#pragma unroll
  for (int j = 0; j < 4; ++j)
#pragma unroll
    for (int r = 0; r < 8; ++r) {
      int pixel = n0 + 16 * wave + r + 8 * lh;
      int d     = 16 * j + ln;
      oh[(pbase + pixel) * C_ + c0 + d] = (_Float16)(oacc[j][r] * lrow[r]);
    }
}

// ---------------------------------------------------------------------------
extern "C" void kernel_launch(void* const* d_in, const int* in_sizes, int n_in,
                              void* d_out, int out_size, void* d_ws, size_t ws_size,
                              hipStream_t stream) {
  (void)in_sizes; (void)n_in; (void)out_size; (void)ws_size;
  const float* x    = (const float*)d_in[0];
  const float* cond = (const float*)d_in[1];
  const float* gqw  = (const float*)d_in[2];
  const float* gqb  = (const float*)d_in[3];
  const float* gkw  = (const float*)d_in[4];
  const float* gkb  = (const float*)d_in[5];
  const float* wq   = (const float*)d_in[6];
  const float* bq   = (const float*)d_in[7];
  const float* wk   = (const float*)d_in[8];
  const float* bk   = (const float*)d_in[9];
  const float* wv   = (const float*)d_in[10];
  const float* bv   = (const float*)d_in[11];
  const float* wo   = (const float*)d_in[12];
  const float* bo   = (const float*)d_in[13];

  char* ws = (char*)d_ws;
  const size_t WSZ = (size_t)C_ * C_ * sizeof(_Float16);          // 128 KB
  const size_t TSZ = (size_t)B_ * C_ * NPIX * sizeof(_Float16);   // 8 MB
  _Float16* wq_h = (_Float16*)(ws + 0 * WSZ);
  _Float16* wk_h = (_Float16*)(ws + 1 * WSZ);
  _Float16* wv_h = (_Float16*)(ws + 2 * WSZ);
  _Float16* wo_h = (_Float16*)(ws + 3 * WSZ);
  char* p = ws + 4 * WSZ;
  _Float16* gnx = (_Float16*)p; p += TSZ;   // pixel-major
  _Float16* gnc = (_Float16*)p; p += TSZ;   // pixel-major
  _Float16* qh  = (_Float16*)p; p += TSZ;   // pixel-major
  _Float16* kh  = (_Float16*)p; p += TSZ;   // pixel-major
  _Float16* vh  = (_Float16*)p; p += TSZ;   // pixel-major
  _Float16* ohb = (_Float16*)p; p += TSZ;   // pixel-major

  const int NW = C_ * C_;
  f32_to_f16_kernel<<<(NW + 255) / 256, 256, 0, stream>>>(wq, wq_h, NW);
  f32_to_f16_kernel<<<(NW + 255) / 256, 256, 0, stream>>>(wk, wk_h, NW);
  f32_to_f16_kernel<<<(NW + 255) / 256, 256, 0, stream>>>(wv, wv_h, NW);
  f32_to_f16_kernel<<<(NW + 255) / 256, 256, 0, stream>>>(wo, wo_h, NW);

  groupnorm_f16_kernel<<<B_ * GROUPS_, 256, 0, stream>>>(x, gqw, gqb, gnx);
  groupnorm_f16_kernel<<<B_ * GROUPS_, 256, 0, stream>>>(cond, gkw, gkb, gnc);

  dim3 ggrid(NPIX / 128, C_ / 64, B_);
  gemm_wmma_kernel<<<ggrid, 256, 0, stream>>>(wq_h, gnx, bq, qh, nullptr, nullptr, 0);
  gemm_wmma_kernel<<<ggrid, 256, 0, stream>>>(wk_h, gnc, bk, kh, nullptr, nullptr, 0);
  gemm_wmma_kernel<<<ggrid, 256, 0, stream>>>(wv_h, gnc, bv, vh, nullptr, nullptr, 0);

  attn_kernel<<<dim3(NPIX / 128, NH_, B_), 256, 0, stream>>>(qh, kh, vh, ohb);

  gemm_wmma_kernel<<<ggrid, 256, 0, stream>>>(wo_h, ohb, bo, nullptr, x,
                                              (float*)d_out, 1);
}
